// Head_15126874817284
// MI455X (gfx1250) — compile-verified, matrix operand order checked by reading, not run on hardware
//
#include <hip/hip_runtime.h>
#include <hip/hip_bf16.h>

#define Bc 4
#define Lc 4096
#define Dc 2048
#define Hc 128

typedef __attribute__((ext_vector_type(16))) __bf16   v16bf;
typedef __attribute__((ext_vector_type(8)))  float    v8f;
typedef __attribute__((ext_vector_type(4)))  unsigned int  u32x4;
typedef __attribute__((ext_vector_type(4)))  float    f32x4;

struct BF16Frag {
    union {
        v16bf          v;
        u32x4          q[2];
        unsigned int   u[8];
        unsigned short s[16];
    };
};

__device__ __forceinline__ unsigned short f2bf(float f) {
    unsigned int u = __builtin_bit_cast(unsigned int, f);
    unsigned int r = u + 0x7FFFu + ((u >> 16) & 1u);   // round-to-nearest-even
    return (unsigned short)(r >> 16);
}

__device__ __forceinline__ v8f wmma_bf16(v16bf a, v16bf b, v8f c) {
    return __builtin_amdgcn_wmma_f32_16x16x32_bf16(
        /*neg_a=*/false, a, /*neg_b=*/false, b,
        /*c_mod=*/(short)0, c, /*reuse_a=*/false, /*reuse_b=*/false);
}

// 16x32 bf16 A/B fragment, K contiguous, `stride` elements between rows/cols.
// Lanes 0-15: idx=lane, K in {0..7}+{16..23}; lanes 16-31: idx=lane-16, K in {8..15}+{24..31}.
__device__ __forceinline__ v16bf load_frag_bf16(const unsigned short* __restrict__ p0,
                                                int stride, int lane) {
    const int idx = lane & 15;
    const int k0  = (lane < 16) ? 0 : 8;
    const unsigned short* p = p0 + (size_t)idx * stride + k0;
    BF16Frag f;
    f.q[0] = *(const u32x4*)(p);
    f.q[1] = *(const u32x4*)(p + 16);
    return f.v;
}

// Same fragment sourced from fp32 memory (fallback path only).
__device__ __forceinline__ v16bf load_frag_f32(const float* __restrict__ p0,
                                               int stride, int lane) {
    const int idx = lane & 15;
    const int k0  = (lane < 16) ? 0 : 8;
    const float* p = p0 + (size_t)idx * stride + k0;
    BF16Frag f;
#pragma unroll
    for (int g = 0; g < 2; ++g) {
        f32x4 a = *(const f32x4*)(p + g * 16);
        f32x4 b = *(const f32x4*)(p + g * 16 + 4);
#pragma unroll
        for (int j = 0; j < 4; ++j) f.s[g * 8 + j]     = f2bf(a[j]);
#pragma unroll
        for (int j = 0; j < 4; ++j) f.s[g * 8 + 4 + j] = f2bf(b[j]);
    }
    return f.v;
}

// ---------------------------------------------------------------------------
// Kernel 0: x [B,L,D] fp32 -> xb [B*L, D] bf16 (8 elems / thread, b128 I/O)
// ---------------------------------------------------------------------------
__global__ __launch_bounds__(256) void xconv_kernel(
    const float* __restrict__ x, unsigned short* __restrict__ xb) {
    size_t i = ((size_t)blockIdx.x * blockDim.x + threadIdx.x) * 8;
    f32x4 a = *(const f32x4*)(x + i);
    f32x4 b = *(const f32x4*)(x + i + 4);
    BF16Frag o;
#pragma unroll
    for (int j = 0; j < 4; ++j) o.s[j]     = f2bf(a[j]);
#pragma unroll
    for (int j = 0; j < 4; ++j) o.s[4 + j] = f2bf(b[j]);
    *(u32x4*)(xb + i) = o.q[0];
}

// ---------------------------------------------------------------------------
// Kernel 1: W [D,H] fp32 -> Wt [H,D] bf16 (x3)
// ---------------------------------------------------------------------------
__global__ __launch_bounds__(256) void wconv_kernel(
    const float* __restrict__ Wq, const float* __restrict__ Wk,
    const float* __restrict__ Wv,
    unsigned short* __restrict__ WtQ, unsigned short* __restrict__ WtK,
    unsigned short* __restrict__ WtV) {
    int i = blockIdx.x * blockDim.x + threadIdx.x;
    if (i >= Dc * Hc) return;
    int d = i / Hc, h = i % Hc;
    size_t o = (size_t)h * Dc + d;
    WtQ[o] = f2bf(Wq[i]);
    WtK[o] = f2bf(Wk[i]);
    WtV[o] = f2bf(Wv[i]);
}

// ---------------------------------------------------------------------------
// Kernel 2: projections, double-buffered over K chunks.
// One block = 16 rows; wave w owns cols [32w, 32w+32) of Q, K and V.
// Q,K -> bf16 row-major [B*L, H]; V -> bf16 transposed [B, H, L].
// ---------------------------------------------------------------------------
template <bool XBF16>
__global__ __launch_bounds__(128) void proj_kernel(
    const float* __restrict__ x,
    const unsigned short* __restrict__ xb,
    const unsigned short* __restrict__ WtQ,
    const unsigned short* __restrict__ WtK,
    const unsigned short* __restrict__ WtV,
    unsigned short* __restrict__ qb,
    unsigned short* __restrict__ kb,
    unsigned short* __restrict__ vT) {
    const int row0 = blockIdx.x * 16;
    const int wave = threadIdx.x >> 5;
    const int lane = threadIdx.x & 31;
    const int col0 = wave * 32;

    const float*          xrow  = x  + (size_t)row0 * Dc;
    const unsigned short* xbrow = xb + (size_t)row0 * Dc;

    auto loadA = [&](int kc) -> v16bf {
        if constexpr (XBF16) return load_frag_bf16(xbrow + kc, Dc, lane);
        else                 return load_frag_f32(xrow + kc, Dc, lane);
    };
    auto loadB = [&](const unsigned short* W, int col, int kc) -> v16bf {
        return load_frag_bf16(W + (size_t)col * Dc + kc, Dc, lane);
    };

    v8f aq0 = {}, aq1 = {}, ak0 = {}, ak1 = {}, av0 = {}, av1 = {};

    // Prime the pipeline.
    v16bf a  = loadA(0);
    v16bf q0 = loadB(WtQ, col0, 0),      q1 = loadB(WtQ, col0 + 16, 0);
    v16bf k0 = loadB(WtK, col0, 0),      k1 = loadB(WtK, col0 + 16, 0);
    v16bf v0 = loadB(WtV, col0, 0),      v1 = loadB(WtV, col0 + 16, 0);

    for (int kc = 0; kc < Dc; kc += 32) {
        const int kn = kc + 32;
        v16bf an = a, q0n = q0, q1n = q1, k0n = k0, k1n = k1, v0n = v0, v1n = v1;
        if (kn < Dc) {
            an  = loadA(kn);
            q0n = loadB(WtQ, col0, kn);  q1n = loadB(WtQ, col0 + 16, kn);
            k0n = loadB(WtK, col0, kn);  k1n = loadB(WtK, col0 + 16, kn);
            v0n = loadB(WtV, col0, kn);  v1n = loadB(WtV, col0 + 16, kn);
        }
        aq0 = wmma_bf16(a, q0, aq0);
        aq1 = wmma_bf16(a, q1, aq1);
        ak0 = wmma_bf16(a, k0, ak0);
        ak1 = wmma_bf16(a, k1, ak1);
        av0 = wmma_bf16(a, v0, av0);
        av1 = wmma_bf16(a, v1, av1);
        a = an; q0 = q0n; q1 = q1n; k0 = k0n; k1 = k1n; v0 = v0n; v1 = v1n;
    }

    // C/D layout: vgpr r, lanes 0-15 -> (M=r, N=lane); lanes 16-31 -> (M=r+8, N=lane-16)
    const int half = lane >> 4;
    const int n    = lane & 15;
#pragma unroll
    for (int r = 0; r < 8; ++r) {
        const int m    = r + 8 * half;
        const int grow = row0 + m;
        const int bb   = grow >> 12;          // / L
        const int ll   = grow & (Lc - 1);
        qb[(size_t)grow * Hc + col0 + n]      = f2bf(aq0[r]);
        qb[(size_t)grow * Hc + col0 + 16 + n] = f2bf(aq1[r]);
        kb[(size_t)grow * Hc + col0 + n]      = f2bf(ak0[r]);
        kb[(size_t)grow * Hc + col0 + 16 + n] = f2bf(ak1[r]);
        vT[((size_t)bb * Hc + col0 + n) * Lc + ll]      = f2bf(av0[r]);
        vT[((size_t)bb * Hc + col0 + 16 + n) * Lc + ll] = f2bf(av1[r]);
    }
}

// ---------------------------------------------------------------------------
// Kernel 3: flash attention. One wave per 16-query tile. Fragment loads are
// issued in batches of 4 distinct registers so VMEM clauses overlap WMMA.
// ---------------------------------------------------------------------------
__global__ __launch_bounds__(32) void attn_kernel(
    const unsigned short* __restrict__ qb,   // [B*L, H] bf16
    const unsigned short* __restrict__ kb,   // [B*L, H] bf16
    const unsigned short* __restrict__ vT,   // [B, H, L] bf16
    float* __restrict__ out) {               // [B*L, H] f32
    __shared__ __align__(16) unsigned short ldsP[16 * 32];

    const int lane = threadIdx.x;
    const int b    = blockIdx.y;
    const int q0   = blockIdx.x * 16;
    const int half = lane >> 4;
    const int n    = lane & 15;
    const float scale = 0.088388347648318447f;   // 1/sqrt(128)

    const unsigned short* qp = qb + ((size_t)b * Lc + q0) * Hc;
    v16bf qf0 = load_frag_bf16(qp,      Hc, lane);
    v16bf qf1 = load_frag_bf16(qp + 32, Hc, lane);
    v16bf qf2 = load_frag_bf16(qp + 64, Hc, lane);
    v16bf qf3 = load_frag_bf16(qp + 96, Hc, lane);

    v8f o[8] = {};
    float mrow[8], lrow[8];
#pragma unroll
    for (int r = 0; r < 8; ++r) { mrow[r] = -1e30f; lrow[r] = 0.f; }

    const int kend = q0 + 16;                // exclusive causal key bound
    for (int kc = 0; kc < kend; kc += 32) {
        const unsigned short* kp = kb + ((size_t)b * Lc + kc) * Hc;
        if (kc + 32 < kend)
            __builtin_prefetch(kp + (size_t)32 * Hc, 0, 1);

        // ---- S = Q K^T: two batches of 4 distinct K fragments ----
        v8f s0, s1;
        {
            v16bf f0 = load_frag_bf16(kp,      Hc, lane);
            v16bf f1 = load_frag_bf16(kp + 32, Hc, lane);
            v16bf f2 = load_frag_bf16(kp + 64, Hc, lane);
            v16bf f3 = load_frag_bf16(kp + 96, Hc, lane);
            v8f t = {};
            t = wmma_bf16(qf0, f0, t);
            t = wmma_bf16(qf1, f1, t);
            t = wmma_bf16(qf2, f2, t);
            t = wmma_bf16(qf3, f3, t);
            s0 = t;
        }
        {
            const unsigned short* kp1 = kp + (size_t)16 * Hc;
            v16bf g0 = load_frag_bf16(kp1,      Hc, lane);
            v16bf g1 = load_frag_bf16(kp1 + 32, Hc, lane);
            v16bf g2 = load_frag_bf16(kp1 + 64, Hc, lane);
            v16bf g3 = load_frag_bf16(kp1 + 96, Hc, lane);
            v8f t = {};
            t = wmma_bf16(qf0, g0, t);
            t = wmma_bf16(qf1, g1, t);
            t = wmma_bf16(qf2, g2, t);
            t = wmma_bf16(qf3, g3, t);
            s1 = t;
        }

        // ---- scale + causal mask + per-row chunk max ----
        float cmax[8];
#pragma unroll
        for (int r = 0; r < 8; ++r) {
            const int qrow = q0 + r + 8 * half;
            float v0 = s0[r] * scale;
            float v1 = s1[r] * scale;
            v0 = ((kc + n)      <= qrow) ? v0 : -1e30f;
            v1 = ((kc + 16 + n) <= qrow) ? v1 : -1e30f;
            s0[r] = v0; s1[r] = v1;
            cmax[r] = fmaxf(v0, v1);
        }
#pragma unroll
        for (int r = 0; r < 8; ++r) {
            cmax[r] = fmaxf(cmax[r], __shfl_xor(cmax[r], 1, 32));
            cmax[r] = fmaxf(cmax[r], __shfl_xor(cmax[r], 2, 32));
            cmax[r] = fmaxf(cmax[r], __shfl_xor(cmax[r], 4, 32));
            cmax[r] = fmaxf(cmax[r], __shfl_xor(cmax[r], 8, 32));
        }

        // ---- online softmax update ----
        float fac[8], rsum[8];
#pragma unroll
        for (int r = 0; r < 8; ++r) {
            float mn = fmaxf(mrow[r], cmax[r]);
            fac[r]  = __expf(mrow[r] - mn);
            mrow[r] = mn;
            float p0 = __expf(s0[r] - mn);
            float p1 = __expf(s1[r] - mn);
            s0[r] = p0; s1[r] = p1;
            rsum[r] = p0 + p1;
        }
#pragma unroll
        for (int r = 0; r < 8; ++r) {
            rsum[r] += __shfl_xor(rsum[r], 1, 32);
            rsum[r] += __shfl_xor(rsum[r], 2, 32);
            rsum[r] += __shfl_xor(rsum[r], 4, 32);
            rsum[r] += __shfl_xor(rsum[r], 8, 32);
            lrow[r] = lrow[r] * fac[r] + rsum[r];
        }
#pragma unroll
        for (int t = 0; t < 8; ++t)
#pragma unroll
            for (int r = 0; r < 8; ++r) o[t][r] *= fac[r];

        // ---- stage P through LDS: C/D layout -> one 16x32 A fragment ----
        __syncthreads();   // read(i-1) -> write(i)
#pragma unroll
        for (int r = 0; r < 8; ++r) {
            const int m = r + 8 * half;
            ldsP[m * 32 + n]      = f2bf(s0[r]);
            ldsP[m * 32 + 16 + n] = f2bf(s1[r]);
        }
        __syncthreads();   // write(i) -> read(i)
        v16bf pa = load_frag_bf16(ldsP, 32, lane);

        // ---- O += P V: two batches of 4 distinct V fragments ----
        const unsigned short* vp = vT + (size_t)b * Hc * Lc + kc;
        {
            v16bf h0 = load_frag_bf16(vp,                     Lc, lane);
            v16bf h1 = load_frag_bf16(vp + (size_t)16 * Lc,   Lc, lane);
            v16bf h2 = load_frag_bf16(vp + (size_t)32 * Lc,   Lc, lane);
            v16bf h3 = load_frag_bf16(vp + (size_t)48 * Lc,   Lc, lane);
            o[0] = wmma_bf16(pa, h0, o[0]);
            o[1] = wmma_bf16(pa, h1, o[1]);
            o[2] = wmma_bf16(pa, h2, o[2]);
            o[3] = wmma_bf16(pa, h3, o[3]);
        }
        {
            v16bf h4 = load_frag_bf16(vp + (size_t)64 * Lc,   Lc, lane);
            v16bf h5 = load_frag_bf16(vp + (size_t)80 * Lc,   Lc, lane);
            v16bf h6 = load_frag_bf16(vp + (size_t)96 * Lc,   Lc, lane);
            v16bf h7 = load_frag_bf16(vp + (size_t)112 * Lc,  Lc, lane);
            o[4] = wmma_bf16(pa, h4, o[4]);
            o[5] = wmma_bf16(pa, h5, o[5]);
            o[6] = wmma_bf16(pa, h6, o[6]);
            o[7] = wmma_bf16(pa, h7, o[7]);
        }
    }

    // ---- epilogue: normalize (rcp once per row) and store fp32 ----
    float rinv[8];
#pragma unroll
    for (int r = 0; r < 8; ++r) rinv[r] = 1.0f / lrow[r];
#pragma unroll
    for (int t = 0; t < 8; ++t) {
#pragma unroll
        for (int r = 0; r < 8; ++r) {
            const int qrow = q0 + r + 8 * half;
            out[((size_t)b * Lc + qrow) * Hc + t * 16 + n] = o[t][r] * rinv[r];
        }
    }
}

// ---------------------------------------------------------------------------
extern "C" void kernel_launch(void* const* d_in, const int* in_sizes, int n_in,
                              void* d_out, int out_size, void* d_ws, size_t ws_size,
                              hipStream_t stream) {
    (void)in_sizes; (void)n_in; (void)out_size;
    const float* x  = (const float*)d_in[0];
    const float* Wq = (const float*)d_in[1];
    const float* Wk = (const float*)d_in[2];
    const float* Wv = (const float*)d_in[3];
    float* out = (float*)d_out;

    // Workspace carve-up (bytes):
    //   qb,kb,vT : B*L*H*2 = 4 MiB each
    //   WtQ/K/V  : D*H*2   = 512 KiB each
    //   xb       : B*L*D*2 = 64 MiB (optional fast path)
    char* ws = (char*)d_ws;
    const size_t szQKV = (size_t)Bc * Lc * Hc * sizeof(unsigned short);
    const size_t szW   = (size_t)Dc * Hc * sizeof(unsigned short);
    const size_t szX   = (size_t)Bc * Lc * Dc * sizeof(unsigned short);
    unsigned short* qb  = (unsigned short*)(ws);
    unsigned short* kb  = (unsigned short*)(ws + szQKV);
    unsigned short* vT  = (unsigned short*)(ws + 2 * szQKV);
    unsigned short* WtQ = (unsigned short*)(ws + 3 * szQKV);
    unsigned short* WtK = (unsigned short*)(ws + 3 * szQKV + szW);
    unsigned short* WtV = (unsigned short*)(ws + 3 * szQKV + 2 * szW);
    unsigned short* xb  = (unsigned short*)(ws + 3 * szQKV + 3 * szW);

    const bool fast = ws_size >= 3 * szQKV + 3 * szW + szX;

    wconv_kernel<<<(Dc * Hc + 255) / 256, 256, 0, stream>>>(Wq, Wk, Wv, WtQ, WtK, WtV);
    if (fast) {
        const size_t nv = (size_t)Bc * Lc * Dc / 8;   // 8 elements per thread
        xconv_kernel<<<(unsigned)(nv / 256), 256, 0, stream>>>(x, xb);
        proj_kernel<true><<<(Bc * Lc) / 16, 128, 0, stream>>>(x, xb, WtQ, WtK, WtV, qb, kb, vT);
    } else {
        proj_kernel<false><<<(Bc * Lc) / 16, 128, 0, stream>>>(x, xb, WtQ, WtK, WtV, qb, kb, vT);
    }
    attn_kernel<<<dim3(Lc / 16, Bc), 32, 0, stream>>>(qb, kb, vT, out);
}
